// MSSMamba_83502754168833
// MI455X (gfx1250) — matrix operand order repaired and checked by reading, not compiled, not run on hardware
//
#include <hip/hip_runtime.h>

#define DEVINL __device__ __forceinline__

// ---- problem constants ----
constexpr int B_    = 2;
constexpr int L_    = 8192;
constexpr int DM    = 256;
constexpr int NH    = 4;
constexpr int HP    = 64;
constexpr int NS    = 8;
constexpr int DIN   = NH * HP;                // 256
constexpr int DPROJ = 2 * DIN + 2 * NS + NH;  // 532
constexpr int ZXN   = 2 * DPROJ;              // 1064
constexpr int ZXP   = 1088;                   // padded: 68 n-tiles (even)
constexpr int ROWS  = B_ * L_;                // 16384
constexpr int NC    = 32;                     // scan chunks
constexpr int CS    = L_ / NC;                // 256 steps per chunk
constexpr int STATE = HP * NS;                // 512 state scalars per (b,h)
constexpr int KT2   = DIN / 32;               // k-tiles of packed y (8)
constexpr int KT3   = 4 * DM / 32;            // k-tiles of packed cat (32)

typedef __attribute__((ext_vector_type(16))) __bf16 v16bf;
typedef __attribute__((ext_vector_type(8)))  float  v8f;

DEVINL __bf16 f2bf(float f) { return (__bf16)f; }  // v_cvt_pk_bf16_f32 path

DEVINL float siluf(float x) { return x / (1.f + __expf(-x)); }

DEVINL v8f wmma_bf16(v16bf a, v16bf b, v8f c) {
  return __builtin_amdgcn_wmma_f32_16x16x32_bf16(false, a, false, b, (short)0,
                                                 c, false, false);
}

// ---------------------------------------------------------------------------
// Pack fp32 weight (K x N row-major) into bf16 WMMA B-fragment tile order:
//   dst[((kt*tn + nt)*32 + lane)*16 + j], k = kt*32 + 16*(j>>3) + 8*(lane>>4) + (j&7),
//   n = nt*16 + (lane&15).  Pad cols (n >= N) are zero.
// ---------------------------------------------------------------------------
__global__ void pack_b_bf16(const float* __restrict__ src, __bf16* __restrict__ dst,
                            int K, int N, int Np) {
  const int idx = blockIdx.x * blockDim.x + threadIdx.x;
  const int tn = Np >> 4;
  if (idx >= (K >> 5) * tn * 512) return;
  const int j    = idx & 15;
  const int lane = (idx >> 4) & 31;
  const int tile = idx >> 9;
  const int kt = tile / tn, nt = tile - kt * tn;
  const int k = (kt << 5) + ((j >> 3) << 4) + ((lane >> 4) << 3) + (j & 7);
  const int n = (nt << 4) + (lane & 15);
  dst[idx] = f2bf((n < N) ? src[(size_t)k * N + n] : 0.f);
}

// ---------------------------------------------------------------------------
// Pack fp32 activations (Mr x K row-major) into bf16 A-fragment tile order:
//   dst[((mt*ktn + kt)*32 + lane)*16 + j],
//   k = kt*32 + 16*(j>>3) + 8*(lane>>4) + (j&7),  m = mt*16 + (lane&15).
// ---------------------------------------------------------------------------
__global__ void pack_a_bf16(const float* __restrict__ src, __bf16* __restrict__ dst,
                            int Mr, int K) {
  const int idx = blockIdx.x * blockDim.x + threadIdx.x;
  const int ktn = K >> 5;
  if (idx >= (Mr >> 4) * ktn * 512) return;
  const int j    = idx & 15;
  const int lane = (idx >> 4) & 31;
  const int tile = idx >> 9;
  const int mt = tile / ktn, kt = tile - mt * ktn;
  const int k = (kt << 5) + ((j >> 3) << 4) + ((lane >> 4) << 3) + (j & 7);
  const int m = (mt << 4) + (lane & 15);
  dst[idx] = f2bf(src[(size_t)m * K + k]);
}

// ---------------------------------------------------------------------------
// WMMA bf16 GEMM:  C[M,N] = A[M,K] * B[K,N], A and B pre-packed fragment tiles.
//   Wave tile: 64x32 (4 m-subtiles x 2 n-subtiles; one A frag feeds 2 WMMAs).
//   SMODE 0: store f32 row-major (ldc), optional coff/rmap.
//   SMODE 1: store silu(acc) as bf16 into packed-A tiles of a downstream GEMM
//            with K-tiles = dstktn (fuses silu + conversion + re-pack).
// ---------------------------------------------------------------------------
template <int SMODE>
__global__ void wmma_gemm(const __bf16* __restrict__ Ap,
                          const __bf16* __restrict__ Bp,
                          float* __restrict__ Cf, __bf16* __restrict__ Cp,
                          int ldc, int dstktn, int coff,
                          const int* __restrict__ rmap,
                          int M, int Np, int K) {
  const int lane = threadIdx.x & 31;
  const int wave = blockIdx.x * (blockDim.x >> 5) + (threadIdx.x >> 5);
  const int tn   = Np >> 4;              // 16-wide n-tiles
  const int tnw  = tn >> 1;              // wave covers two n-tiles
  const int tm   = wave / tnw;
  const int nt0  = (wave - tm * tnw) << 1;
  const int m0   = tm << 6;
  if (m0 >= M) return;        // wave-uniform: EXEC stays all-ones for WMMA
  const int ktn  = K >> 5;

  v8f acc[4][2] = {};

  unsigned boff = ((unsigned)nt0 * 32u + (unsigned)lane) << 4;   // elements
  const unsigned bstep = (unsigned)tn << 9;                      // tn*512
  const unsigned abase = (((unsigned)(m0 >> 4) * (unsigned)ktn) * 32u +
                          (unsigned)lane) << 4;
  const unsigned atile = (unsigned)ktn << 9;                     // m-tile stride

  for (int kt = 0; kt < ktn; ++kt) {
    const v16bf b0 = *(const v16bf*)(Bp + boff);
    const v16bf b1 = *(const v16bf*)(Bp + boff + 512u);
    boff += bstep;
    const unsigned ak = abase + ((unsigned)kt << 9);
#pragma unroll
    for (int i = 0; i < 4; ++i) {
      const v16bf a = *(const v16bf*)(Ap + ak + (unsigned)i * atile);
      acc[i][0] = wmma_bf16(a, b0, acc[i][0]);
      acc[i][1] = wmma_bf16(a, b1, acc[i][1]);
    }
  }

  const int colb = coff + (nt0 << 4) + (lane & 15);
#pragma unroll
  for (int i = 0; i < 4; ++i) {
    const int mr = m0 + 16 * i + ((lane >> 4) << 3);
#pragma unroll
    for (int h2 = 0; h2 < 2; ++h2) {
      const int col = colb + 16 * h2;
#pragma unroll
      for (int r = 0; r < 8; ++r) {
        int row = mr + r;
        if (rmap) {              // v2h scatter: row -> b*L + v2h[b][l]
          const int bb = row / L_;
          row = bb * L_ + rmap[row];
        }
        if (SMODE == 0) {
          Cf[(size_t)row * (unsigned)ldc + (unsigned)col] = acc[i][h2][r];
        } else {
          const int kcol  = col & 31;
          const int kt3   = col >> 5;
          const int lane3 = (row & 15) | (((kcol >> 3) & 1) << 4);
          const int j3    = (kcol & 7) | ((kcol >> 4) << 3);
          const size_t pidx =
              (((size_t)(row >> 4) * dstktn + kt3) * 32 + lane3) * 16 + j3;
          Cp[pidx] = f2bf(siluf(acc[i][h2][r]));
        }
      }
    }
  }
}

// ---------------------------------------------------------------------------
// Chunked selective scan (3 phases).  Recurrence per (b,h,p,n):
//   s = exp(dt*A)*s + dt*silu(B_n)*silu(x_hp)
// dA is scalar per (b,h,t) so chunks combine affinely:
//   s_end = (prod dA)*s_in + s_local
// pass 1: local scan from 0 -> chunk summary (state, prod dA)
// mid   : 32-step scan of summaries -> per-chunk prefix states
// pass 2: replay with prefix, emit gated y in packed A-fragment bf16 order.
// Block = 512 threads = one (b,h); thread = (p = tid>>3, n = tid&7).
// ---------------------------------------------------------------------------
__global__ void scan_chunk(const float* __restrict__ zx, int col0, int rev,
                           const float* __restrict__ dt_bias,
                           const float* __restrict__ A_log,
                           const float* __restrict__ Dvec,
                           float* __restrict__ sumS, float* __restrict__ sumP,
                           const float* __restrict__ prefS,
                           __bf16* __restrict__ ypack, int final_pass) {
  const int c   = blockIdx.x;
  const int bh  = blockIdx.y;
  const int b   = bh / NH;
  const int h   = bh - b * NH;
  const int tid = threadIdx.x;
  const int pp  = tid >> 3;
  const int nn  = tid & 7;

  const float Ahf  = -__expf(A_log[h]);
  const float bias = dt_bias[h];
  const float Dh   = Dvec[h];

  const size_t sidx = ((size_t)bh * NC + c) * STATE + tid;
  float s  = final_pass ? prefS[sidx] : 0.f;
  float Pp = 1.f;

  __shared__ float shx[HP];
  __shared__ float shz[HP];
  __shared__ float shB[NS];
  __shared__ float shC[NS];
  __shared__ float shdt[1];

  const int col  = h * HP + pp;             // K index in out-proj GEMM
  const int kcol = col & 31;
  const int kt2  = col >> 5;
  const int jj   = (kcol & 7) | ((kcol >> 4) << 3);

  const int u0 = c * CS;
  for (int uu = 0; uu < CS; ++uu) {
    const int u = u0 + uu;
    const int t = rev ? (L_ - 1 - u) : u;
    const size_t base = (size_t)(b * L_ + t) * ZXP + col0;

    if (tid < HP)                 shx[tid]           = zx[base + DIN + h * HP + tid];
    else if (tid < HP + NS)       shB[tid - HP]      = zx[base + 2 * DIN + (tid - HP)];
    else if (tid < HP + 2 * NS)   shC[tid - HP - NS] = zx[base + 2 * DIN + NS + (tid - HP - NS)];
    else if (tid == HP + 2 * NS)  shdt[0]            = zx[base + 2 * DIN + 2 * NS + h];
    else if (tid >= 128 && tid < 128 + HP) {
      if (final_pass)             shz[tid - 128]     = zx[base + h * HP + (tid - 128)];
    }
    __syncthreads();

    const float dtr = shdt[0] + bias;
    const float dt  = (dtr > 20.f) ? dtr : log1pf(__expf(dtr));   // softplus
    const float dA  = __expf(dt * Ahf);
    const float xv  = siluf(shx[pp]);
    const float Bn  = siluf(shB[nn]);
    s = dA * s + dt * Bn * xv;

    if (final_pass) {
      float yt = s * siluf(shC[nn]);
      yt += __shfl_xor(yt, 1);          // reduce over the 8 n-lanes (wave32)
      yt += __shfl_xor(yt, 2);
      yt += __shfl_xor(yt, 4);
      if (nn == 0) {
        const float z = shz[pp];
        const float y = (yt + Dh * xv) * siluf(z);
        const int row   = b * L_ + t;
        const int lane2 = (row & 15) | (((kcol >> 3) & 1) << 4);
        const size_t pidx =
            ((((size_t)(row >> 4)) * KT2 + kt2) * 32 + lane2) * 16 + jj;
        ypack[pidx] = f2bf(y);
      }
    } else {
      Pp *= dA;
    }
    __syncthreads();
  }

  if (!final_pass) {
    sumS[sidx] = s;
    if (tid == 0) sumP[(size_t)bh * NC + c] = Pp;
  }
}

__global__ void scan_mid(const float* __restrict__ sumS,
                         const float* __restrict__ sumP,
                         float* __restrict__ prefS) {
  const int bh  = blockIdx.x;
  const int tid = threadIdx.x;
  float s = 0.f;
  for (int c = 0; c < NC; ++c) {
    const size_t idx = ((size_t)bh * NC + c) * STATE + tid;
    prefS[idx] = s;                         // exclusive prefix state
    s = sumP[(size_t)bh * NC + c] * s + sumS[idx];
  }
}

// ---------------------------------------------------------------------------
extern "C" void kernel_launch(void* const* d_in, const int* in_sizes, int n_in,
                              void* d_out, int out_size, void* d_ws, size_t ws_size,
                              hipStream_t stream) {
  (void)in_sizes; (void)n_in; (void)out_size; (void)ws_size;

  const float* x_H    = (const float*)d_in[0];
  const float* x_V    = (const float*)d_in[1];
  const float* W_in_H = (const float*)d_in[2];
  const float* W_in_V = (const float*)d_in[3];
  const float* W_out_dir[4] = {(const float*)d_in[4], (const float*)d_in[5],
                               (const float*)d_in[6], (const float*)d_in[7]};
  const float* W_out  = (const float*)d_in[8];
  const float* dtb[4]  = {(const float*)d_in[9],  (const float*)d_in[12],
                          (const float*)d_in[15], (const float*)d_in[18]};
  const float* alog[4] = {(const float*)d_in[10], (const float*)d_in[13],
                          (const float*)d_in[16], (const float*)d_in[19]};
  const float* Dd[4]   = {(const float*)d_in[11], (const float*)d_in[14],
                          (const float*)d_in[17], (const float*)d_in[20]};
  const int* v2h = (const int*)d_in[21];

  // ---- workspace carve-out ----
  char* wp = (char*)d_ws;
  auto take = [&](size_t bytes) {
    char* q = wp;
    wp += (bytes + 255) & ~(size_t)255;
    return q;
  };
  float*  zxH   = (float*)take((size_t)ROWS * ZXP * 4);
  float*  zxV   = (float*)take((size_t)ROWS * ZXP * 4);
  __bf16* xpH   = (__bf16*)take((size_t)ROWS * DM * 2);     // packed A of x_H
  __bf16* xpV   = (__bf16*)take((size_t)ROWS * DM * 2);
  __bf16* catp  = (__bf16*)take((size_t)ROWS * 4 * DM * 2); // packed A of silu(cat)
  __bf16* yb[4];
  for (int d = 0; d < 4; ++d) yb[d] = (__bf16*)take((size_t)ROWS * DIN * 2);
  __bf16* WinHb = (__bf16*)take((size_t)DM * ZXP * 2);
  __bf16* WinVb = (__bf16*)take((size_t)DM * ZXP * 2);
  __bf16* WoutDirB[4];
  for (int d = 0; d < 4; ++d) WoutDirB[d] = (__bf16*)take((size_t)DIN * DM * 2);
  __bf16* Woutb = (__bf16*)take((size_t)4 * DM * DM * 2);
  const size_t sumElems = (size_t)B_ * NH * NC * STATE;     // per direction
  float* sumS  = (float*)take(4 * sumElems * 4);
  float* prefS = (float*)take(4 * sumElems * 4);
  float* sumP  = (float*)take((size_t)4 * B_ * NH * NC * 4);

  // ---- 1) packs: weights -> B tiles, x -> A tiles (convert once) ----
  {
    int t = (DM / 32) * (ZXP / 16) * 512;
    pack_b_bf16<<<(t + 255) / 256, 256, 0, stream>>>(W_in_H, WinHb, DM, ZXN, ZXP);
    pack_b_bf16<<<(t + 255) / 256, 256, 0, stream>>>(W_in_V, WinVb, DM, ZXN, ZXP);
    t = (DIN / 32) * (DM / 16) * 512;
    for (int d = 0; d < 4; ++d)
      pack_b_bf16<<<(t + 255) / 256, 256, 0, stream>>>(W_out_dir[d], WoutDirB[d],
                                                       DIN, DM, DM);
    t = (4 * DM / 32) * (DM / 16) * 512;
    pack_b_bf16<<<(t + 255) / 256, 256, 0, stream>>>(W_out, Woutb, 4 * DM, DM, DM);
    t = (ROWS / 16) * (DM / 32) * 512;
    pack_a_bf16<<<(t + 255) / 256, 256, 0, stream>>>(x_H, xpH, ROWS, DM);
    pack_a_bf16<<<(t + 255) / 256, 256, 0, stream>>>(x_V, xpV, ROWS, DM);
  }

  auto grid_for = [](int M, int Np) {
    const int waves = (M / 64) * (Np / 32);   // 64x32 wave tiles
    return (waves + 7) / 8;                   // 8 waves per 256-thread block
  };

  // ---- 2) input projections ----
  wmma_gemm<0><<<grid_for(ROWS, ZXP), 256, 0, stream>>>(
      xpH, WinHb, zxH, nullptr, ZXP, 0, 0, nullptr, ROWS, ZXP, DM);
  wmma_gemm<0><<<grid_for(ROWS, ZXP), 256, 0, stream>>>(
      xpV, WinVb, zxV, nullptr, ZXP, 0, 0, nullptr, ROWS, ZXP, DM);

  // ---- 3) four chunked scans (HF, HB, VF, VB) ----
  const float* zxs[4] = {zxH, zxH, zxV, zxV};
  const int col0[4]   = {0, DPROJ, 0, DPROJ};
  const int rev[4]    = {0, 1, 0, 1};
  for (int d = 0; d < 4; ++d) {
    float* sS = sumS  + d * sumElems;
    float* pS = prefS + d * sumElems;
    float* sP = sumP  + (size_t)d * B_ * NH * NC;
    scan_chunk<<<dim3(NC, B_ * NH), STATE, 0, stream>>>(
        zxs[d], col0[d], rev[d], dtb[d], alog[d], Dd[d], sS, sP, pS, yb[d], 0);
    scan_mid<<<B_ * NH, STATE, 0, stream>>>(sS, sP, pS);
    scan_chunk<<<dim3(NC, B_ * NH), STATE, 0, stream>>>(
        zxs[d], col0[d], rev[d], dtb[d], alog[d], Dd[d], sS, sP, pS, yb[d], 1);
  }

  // ---- 4) out-projections, fused silu+bf16 re-pack into catp,
  //         V directions scattered through v2h on store
  //         (reference swaps F/B weights) ----
  wmma_gemm<1><<<grid_for(ROWS, DM), 256, 0, stream>>>(
      yb[0], WoutDirB[1], nullptr, catp, 0, KT3, 0 * DM, nullptr,
      ROWS, DM, DIN);                                    // y_HF @ W_out_HB
  wmma_gemm<1><<<grid_for(ROWS, DM), 256, 0, stream>>>(
      yb[1], WoutDirB[0], nullptr, catp, 0, KT3, 1 * DM, nullptr,
      ROWS, DM, DIN);                                    // y_HB @ W_out_HF
  wmma_gemm<1><<<grid_for(ROWS, DM), 256, 0, stream>>>(
      yb[2], WoutDirB[3], nullptr, catp, 0, KT3, 2 * DM, v2h,
      ROWS, DM, DIN);                                    // y_VF @ W_out_VB
  wmma_gemm<1><<<grid_for(ROWS, DM), 256, 0, stream>>>(
      yb[3], WoutDirB[2], nullptr, catp, 0, KT3, 3 * DM, v2h,
      ROWS, DM, DIN);                                    // y_VB @ W_out_VF

  // ---- 5) final: silu(cat) @ W_out -> d_out ----
  wmma_gemm<0><<<grid_for(ROWS, DM), 256, 0, stream>>>(
      catp, Woutb, (float*)d_out, nullptr, DM, 0, 0, nullptr,
      ROWS, DM, 4 * DM);
}
